// GNN_SAT_13597866459551
// MI455X (gfx1250) — matrix operand order repaired and checked by reading, not compiled, not run on hardware
//
#include <hip/hip_runtime.h>
#include <cstddef>
#include <cstdint>

// ---------------------------------------------------------------------------
// Problem constants (match reference setup_inputs)
// ---------------------------------------------------------------------------
constexpr int D    = 128;
constexpr int BB   = 32;            // graphs in batch
constexpr int NV   = 500;
constexpr int NL   = BB * 2 * NV;   // 32000 literals
constexpr int NC_G = 2000;
constexpr int NC   = BB * NC_G;     // 64000 clauses
constexpr int KE   = 3;             // literals per clause
constexpr int NE   = NC * KE;       // 192000 edges
constexpr int NUM_ITERS = 4;        // reference num_iters (host must know launch count)
constexpr int G4D  = 4 * D;         // 512 gate outputs

// ---------------------------------------------------------------------------
// Types for WMMA
// ---------------------------------------------------------------------------
typedef __attribute__((ext_vector_type(16))) __bf16        v16bf;
typedef __attribute__((ext_vector_type(8)))  float         v8f;
typedef __attribute__((ext_vector_type(4)))  unsigned int  u32x4;

union FragU { u32x4 u[2]; v16bf v; };

__device__ __forceinline__ unsigned short f2bf(float f) {
  unsigned int u = __float_as_uint(f);
  unsigned int r = u + 0x7FFFu + ((u >> 16) & 1u);   // round-to-nearest-even
  return (unsigned short)(r >> 16);
}
__device__ __forceinline__ float sigmf(float x) {
  return 1.0f / (1.0f + __expf(-x));
}

// ---------------------------------------------------------------------------
// Init / pack kernels
// ---------------------------------------------------------------------------
__global__ void init_lits_kernel(const float* __restrict__ x_unk,
                                 float* __restrict__ h_l, float* __restrict__ c_l) {
  int idx = blockIdx.x * blockDim.x + threadIdx.x;
  if (idx < NL * D) { h_l[idx] = x_unk[idx]; c_l[idx] = 0.0f; }
}

__global__ void init_clauses_kernel(const float* __restrict__ Cw, const float* __restrict__ Cb,
                                    float* __restrict__ h_c, float* __restrict__ c_c) {
  int idx = blockIdx.x * blockDim.x + threadIdx.x;
  if (idx < NC * D) {
    int d = idx & (D - 1);
    h_c[idx] = Cw[d] + Cb[d];
    c_c[idx] = 0.0f;
  }
}

// Pack W = [Wih | Whh] row-major [512, Ktot] into bf16; bias = bih + bhh.
__global__ void pack_weights_kernel(const float* __restrict__ Wih, const float* __restrict__ Whh,
                                    const float* __restrict__ bih, const float* __restrict__ bhh,
                                    unsigned short* __restrict__ Wpk, float* __restrict__ bpk,
                                    int Kx, int Kh) {
  int Ktot = Kx + Kh;
  int idx = blockIdx.x * blockDim.x + threadIdx.x;
  if (idx >= G4D * Ktot) return;
  int n = idx / Ktot;
  int k = idx - n * Ktot;
  float w = (k < Kx) ? Wih[(size_t)n * Kx + k] : Whh[(size_t)n * Kh + (k - Kx)];
  Wpk[idx] = f2bf(w);
  if (k == 0) bpk[n] = bih[n] + bhh[n];
}

// ---------------------------------------------------------------------------
// CSR build for literal <- clause gather (replaces fp32 atomic scatter;
// sorted rows give a deterministic, fixed-order summation)
// ---------------------------------------------------------------------------
__global__ void zero_int_kernel(int* __restrict__ p, int n) {
  int idx = blockIdx.x * blockDim.x + threadIdx.x;
  if (idx < n) p[idx] = 0;
}

__global__ void count_deg_kernel(const int* __restrict__ lit_idx, int* __restrict__ deg) {
  int e = blockIdx.x * blockDim.x + threadIdx.x;
  if (e < NE) atomicAdd(&deg[lit_idx[e]], 1);
}

// Single-block exclusive scan: row_ptr[0..NL], 1024 threads x 32 chunk.
__global__ __launch_bounds__(1024) void scan_kernel(const int* __restrict__ deg,
                                                    int* __restrict__ row_ptr) {
  constexpr int CH = 32;             // 1024 * 32 = 32768 >= NL
  __shared__ int part[1024];
  int t = threadIdx.x;
  int s = 0;
  for (int i = 0; i < CH; ++i) {
    int idx = t * CH + i;
    if (idx < NL) s += deg[idx];
  }
  part[t] = s;
  __syncthreads();
  for (int off = 1; off < 1024; off <<= 1) {   // inclusive scan in LDS
    int v = (t >= off) ? part[t - off] : 0;
    __syncthreads();
    part[t] += v;
    __syncthreads();
  }
  int run = (t == 0) ? 0 : part[t - 1];
  for (int i = 0; i < CH; ++i) {
    int idx = t * CH + i;
    if (idx < NL) { row_ptr[idx] = run; run += deg[idx]; }
  }
  if (t == 0) row_ptr[NL] = part[1023];
}

__global__ void copy_int_kernel(const int* __restrict__ src, int* __restrict__ dst, int n) {
  int idx = blockIdx.x * blockDim.x + threadIdx.x;
  if (idx < n) dst[idx] = src[idx];
}

__global__ void fill_csr_kernel(const int* __restrict__ lit_idx,
                                int* __restrict__ cursor, int* __restrict__ col_clause) {
  int e = blockIdx.x * blockDim.x + threadIdx.x;
  if (e >= NE) return;
  int slot = atomicAdd(&cursor[lit_idx[e]], 1);
  col_clause[slot] = e / 3;          // clause id of edge e
}

// Insertion-sort each row (avg degree ~6) -> deterministic gather order.
__global__ void sort_rows_kernel(const int* __restrict__ row_ptr, int* __restrict__ col) {
  int l = blockIdx.x * blockDim.x + threadIdx.x;
  if (l >= NL) return;
  int b0 = row_ptr[l], b1 = row_ptr[l + 1];
  for (int i = b0 + 1; i < b1; ++i) {
    int key = col[i];
    int j = i - 1;
    while (j >= b0 && col[j] > key) { col[j + 1] = col[j]; --j; }
    col[j + 1] = key;
  }
}

// ---------------------------------------------------------------------------
// A-matrix builders
// ---------------------------------------------------------------------------
// A_c[c, 0:128]   = bf16( sum_{k<3} h_l[lit_idx[3c+k]] )   (lit->clause segment sum)
// A_c[c, 128:256] = bf16( h_c[c] )
__global__ void build_Ac_kernel(const float* __restrict__ h_l, const float* __restrict__ h_c,
                                const int* __restrict__ lit_idx,
                                unsigned short* __restrict__ Ac) {
  int idx = blockIdx.x * blockDim.x + threadIdx.x;
  if (idx >= NC * D) return;
  int c = idx >> 7;
  int d = idx & (D - 1);
  int e = 3 * c;
  float s = h_l[(size_t)lit_idx[e]     * D + d]
          + h_l[(size_t)lit_idx[e + 1] * D + d]
          + h_l[(size_t)lit_idx[e + 2] * D + d];
  Ac[(size_t)c * 256 + d]       = f2bf(s);
  Ac[(size_t)c * 256 + 128 + d] = f2bf(h_c[idx]);
}

// A_l[l, 0:128]   = bf16( sum over CSR row l of h_c[clause] )  (fused gather)
// A_l[l, 128:256] = bf16( h_l[flip_perm[l]] )
// A_l[l, 256:384] = bf16( h_l[l] )
__global__ void build_Al_kernel(const float* __restrict__ h_c, const float* __restrict__ h_l,
                                const int* __restrict__ row_ptr, const int* __restrict__ col,
                                const int* __restrict__ flip_perm,
                                unsigned short* __restrict__ Al) {
  int idx = blockIdx.x * blockDim.x + threadIdx.x;
  if (idx >= NL * D) return;
  int l = idx >> 7;
  int d = idx & (D - 1);
  float s = 0.0f;
  int b0 = row_ptr[l], b1 = row_ptr[l + 1];
  for (int j = b0; j < b1; ++j) s += h_c[(size_t)col[j] * D + d];
  Al[(size_t)l * 384 + d]       = f2bf(s);
  Al[(size_t)l * 384 + 128 + d] = f2bf(h_l[(size_t)flip_perm[l] * D + d]);
  Al[(size_t)l * 384 + 256 + d] = f2bf(h_l[idx]);
}

// ---------------------------------------------------------------------------
// Fused GEMM (bf16 WMMA, fp32 acc) + LSTM cell update.
// A: [M, Ktot] bf16 row-major. W: [512, Ktot] bf16 row-major (B = W^T).
// Each wave computes TWO 16-row m-tiles x one 16-col d-tile for all four
// gates (8 accumulators); B fragments are reused across the two m-tiles.
// block = (32, 8): wave y handles d-tile y; grid.x = M/32.
// ---------------------------------------------------------------------------
__global__ __launch_bounds__(256) void gemm_lstm_kernel(
    const unsigned short* __restrict__ A,
    const unsigned short* __restrict__ W,
    const float* __restrict__ bias,
    float* __restrict__ cbuf,   // [M, 128] in/out
    float* __restrict__ hbuf,   // [M, 128] out
    int Ktot)
{
  const int lane = threadIdx.x;        // 0..31
  const int colg = lane & 15;          // column within 16-wide tile
  const int hi   = lane >> 4;          // half-wave select
  const int d0   = threadIdx.y << 4;   // 0..112
  const int m0   = blockIdx.x << 5;    // two m-tiles per wave

  v8f acc0[4] = {};   // m-tile 0: gates i, f, g, o
  v8f acc1[4] = {};   // m-tile 1

  // A fragment: lane holds row; lanes 0-15 K {0..15}, lanes 16-31 K {16..31}
  const unsigned short* arow0 = A + (size_t)(m0 + colg) * Ktot + (hi << 4);
  const unsigned short* arow1 = arow0 + (size_t)16 * Ktot;
  // B fragment: lane holds output column (gate*128 + d0 + colg); 16 contiguous K values
  const unsigned short* w0 = W + (size_t)(0 * D + d0 + colg) * Ktot + (hi << 4);
  const unsigned short* w1 = W + (size_t)(1 * D + d0 + colg) * Ktot + (hi << 4);
  const unsigned short* w2 = W + (size_t)(2 * D + d0 + colg) * Ktot + (hi << 4);
  const unsigned short* w3 = W + (size_t)(3 * D + d0 + colg) * Ktot + (hi << 4);

  for (int k0 = 0; k0 < Ktot; k0 += 32) {
    FragU a0, a1;
    a0.u[0] = *(const u32x4*)(arow0 + k0);
    a0.u[1] = *(const u32x4*)(arow0 + k0 + 8);
    a1.u[0] = *(const u32x4*)(arow1 + k0);
    a1.u[1] = *(const u32x4*)(arow1 + k0 + 8);

    FragU b;
    b.u[0] = *(const u32x4*)(w0 + k0); b.u[1] = *(const u32x4*)(w0 + k0 + 8);
    acc0[0] = __builtin_amdgcn_wmma_f32_16x16x32_bf16(false, a0.v, false, b.v, (short)0, acc0[0], false, false);
    acc1[0] = __builtin_amdgcn_wmma_f32_16x16x32_bf16(false, a1.v, false, b.v, (short)0, acc1[0], false, false);
    b.u[0] = *(const u32x4*)(w1 + k0); b.u[1] = *(const u32x4*)(w1 + k0 + 8);
    acc0[1] = __builtin_amdgcn_wmma_f32_16x16x32_bf16(false, a0.v, false, b.v, (short)0, acc0[1], false, false);
    acc1[1] = __builtin_amdgcn_wmma_f32_16x16x32_bf16(false, a1.v, false, b.v, (short)0, acc1[1], false, false);
    b.u[0] = *(const u32x4*)(w2 + k0); b.u[1] = *(const u32x4*)(w2 + k0 + 8);
    acc0[2] = __builtin_amdgcn_wmma_f32_16x16x32_bf16(false, a0.v, false, b.v, (short)0, acc0[2], false, false);
    acc1[2] = __builtin_amdgcn_wmma_f32_16x16x32_bf16(false, a1.v, false, b.v, (short)0, acc1[2], false, false);
    b.u[0] = *(const u32x4*)(w3 + k0); b.u[1] = *(const u32x4*)(w3 + k0 + 8);
    acc0[3] = __builtin_amdgcn_wmma_f32_16x16x32_bf16(false, a0.v, false, b.v, (short)0, acc0[3], false, false);
    acc1[3] = __builtin_amdgcn_wmma_f32_16x16x32_bf16(false, a1.v, false, b.v, (short)0, acc1[3], false, false);
  }

  const float bi = bias[0 * D + d0 + colg];
  const float bf = bias[1 * D + d0 + colg];
  const float bg = bias[2 * D + d0 + colg];
  const float bo = bias[3 * D + d0 + colg];

  // C layout: VGPR r, lanes 0-15 -> (M=r, N=colg); lanes 16-31 -> (M=8+r, N=colg)
  const int rowb = m0 + (hi << 3);
  #pragma unroll
  for (int r = 0; r < 8; ++r) {
    size_t off = (size_t)(rowb + r) * D + d0 + colg;
    float iv = sigmf(acc0[0][r] + bi);
    float fv = sigmf(acc0[1][r] + bf);
    float gv = tanhf(acc0[2][r] + bg);
    float ov = sigmf(acc0[3][r] + bo);
    float cn = fv * cbuf[off] + iv * gv;
    cbuf[off] = cn;
    hbuf[off] = ov * tanhf(cn);
  }
  #pragma unroll
  for (int r = 0; r < 8; ++r) {
    size_t off = (size_t)(rowb + 16 + r) * D + d0 + colg;
    float iv = sigmf(acc1[0][r] + bi);
    float fv = sigmf(acc1[1][r] + bf);
    float gv = tanhf(acc1[2][r] + bg);
    float ov = sigmf(acc1[3][r] + bo);
    float cn = fv * cbuf[off] + iv * gv;
    cbuf[off] = cn;
    hbuf[off] = ov * tanhf(cn);
  }
}

// ---------------------------------------------------------------------------
// Readout
// ---------------------------------------------------------------------------
__global__ __launch_bounds__(256) void votes_kernel(
    const float* __restrict__ h_l, const float* __restrict__ out_w,
    const float* __restrict__ out_b, float* __restrict__ votes_out)
{
  int wid  = (blockIdx.x * blockDim.x + threadIdx.x) >> 5;   // one wave per literal
  int lane = threadIdx.x & 31;
  if (wid >= NL) return;
  float acc = 0.0f;
  #pragma unroll
  for (int d = lane; d < D; d += 32) acc += h_l[(size_t)wid * D + d] * out_w[d];
  #pragma unroll
  for (int off = 16; off > 0; off >>= 1) acc += __shfl_xor(acc, off, 32);
  if (lane == 0) votes_out[wid] = acc + out_b[0];
}

// Deterministic per-graph mean: one block per graph, fixed-order tree reduce.
__global__ __launch_bounds__(256) void batch_reduce_kernel(
    const float* __restrict__ votes, const int* __restrict__ l_batch,
    float* __restrict__ out)
{
  __shared__ float ssum[256];
  __shared__ float scnt[256];
  int b = blockIdx.x;
  int t = threadIdx.x;
  float s = 0.0f, c = 0.0f;
  for (int l = t; l < NL; l += 256) {
    if (l_batch[l] == b) { s += votes[l]; c += 1.0f; }
  }
  ssum[t] = s; scnt[t] = c;
  __syncthreads();
  for (int off = 128; off > 0; off >>= 1) {
    if (t < off) { ssum[t] += ssum[t + off]; scnt[t] += scnt[t + off]; }
    __syncthreads();
  }
  if (t == 0) out[b] = ssum[0] / scnt[0];
}

__global__ void copy_kernel(const float* __restrict__ src, float* __restrict__ dst, int n) {
  int idx = blockIdx.x * blockDim.x + threadIdx.x;
  if (idx < n) dst[idx] = src[idx];
}

// ---------------------------------------------------------------------------
// Host driver
// ---------------------------------------------------------------------------
extern "C" void kernel_launch(void* const* d_in, const int* in_sizes, int n_in,
                              void* d_out, int out_size, void* d_ws, size_t ws_size,
                              hipStream_t stream) {
  (void)in_sizes; (void)n_in; (void)out_size; (void)ws_size;

  const float* x_unk   = (const float*)d_in[0];
  const float* C_w     = (const float*)d_in[1];
  const float* C_b     = (const float*)d_in[2];
  const float* Wih_lc  = (const float*)d_in[3];
  const float* Whh_lc  = (const float*)d_in[4];
  const float* bih_lc  = (const float*)d_in[5];
  const float* bhh_lc  = (const float*)d_in[6];
  const float* Wih_cl  = (const float*)d_in[7];
  const float* Whh_cl  = (const float*)d_in[8];
  const float* bih_cl  = (const float*)d_in[9];
  const float* bhh_cl  = (const float*)d_in[10];
  const float* out_w   = (const float*)d_in[11];
  const float* out_b   = (const float*)d_in[12];
  const int*   lit_idx = (const int*)d_in[13];
  /* d_in[14] = clause_idx: implicit (edge e belongs to clause e/3) */
  const int*   l_batch = (const int*)d_in[15];
  const int*   flip    = (const int*)d_in[16];
  /* d_in[17] = num_iters: device scalar; reference constant 4, hardcoded */

  char* p = (char*)d_ws;
  auto take = [&](size_t bytes) -> void* {
    void* r = (void*)p;
    p += (bytes + 255) & ~(size_t)255;
    return r;
  };
  float* h_l   = (float*)take((size_t)NL * D * 4);
  float* c_l   = (float*)take((size_t)NL * D * 4);
  float* h_c   = (float*)take((size_t)NC * D * 4);
  float* c_c   = (float*)take((size_t)NC * D * 4);
  unsigned short* Abuf   = (unsigned short*)take((size_t)NC * 256 * 2); // also holds NL*384
  unsigned short* Wpk_lc = (unsigned short*)take((size_t)G4D * 256 * 2);
  unsigned short* Wpk_cl = (unsigned short*)take((size_t)G4D * 384 * 2);
  float* b_lc = (float*)take((size_t)G4D * 4);
  float* b_cl = (float*)take((size_t)G4D * 4);
  int* deg     = (int*)take((size_t)NL * 4);
  int* row_ptr = (int*)take((size_t)(NL + 1) * 4);
  int* cursor  = (int*)take((size_t)NL * 4);
  int* col     = (int*)take((size_t)NE * 4);

  auto g1 = [](int n) { return dim3((unsigned)((n + 255) / 256)); };
  dim3 gblk(32, 8);

  // --- init state + pack weights ---
  init_lits_kernel<<<g1(NL * D), 256, 0, stream>>>(x_unk, h_l, c_l);
  init_clauses_kernel<<<g1(NC * D), 256, 0, stream>>>(C_w, C_b, h_c, c_c);
  pack_weights_kernel<<<g1(G4D * 256), 256, 0, stream>>>(Wih_lc, Whh_lc, bih_lc, bhh_lc,
                                                         Wpk_lc, b_lc, D, D);
  pack_weights_kernel<<<g1(G4D * 384), 256, 0, stream>>>(Wih_cl, Whh_cl, bih_cl, bhh_cl,
                                                         Wpk_cl, b_cl, 2 * D, D);

  // --- build sorted CSR (literal -> clauses) once per call ---
  zero_int_kernel<<<g1(NL), 256, 0, stream>>>(deg, NL);
  count_deg_kernel<<<g1(NE), 256, 0, stream>>>(lit_idx, deg);
  scan_kernel<<<1, 1024, 0, stream>>>(deg, row_ptr);
  copy_int_kernel<<<g1(NL), 256, 0, stream>>>(row_ptr, cursor, NL);
  fill_csr_kernel<<<g1(NE), 256, 0, stream>>>(lit_idx, cursor, col);
  sort_rows_kernel<<<g1(NL), 256, 0, stream>>>(row_ptr, col);

  // --- message-passing iterations ---
  for (int it = 0; it < NUM_ITERS; ++it) {
    build_Ac_kernel<<<g1(NC * D), 256, 0, stream>>>(h_l, h_c, lit_idx, Abuf);
    gemm_lstm_kernel<<<dim3(NC / 32), gblk, 0, stream>>>(Abuf, Wpk_lc, b_lc, c_c, h_c, 2 * D);
    build_Al_kernel<<<g1(NL * D), 256, 0, stream>>>(h_c, h_l, row_ptr, col, flip, Abuf);
    gemm_lstm_kernel<<<dim3(NL / 32), gblk, 0, stream>>>(Abuf, Wpk_cl, b_cl, c_l, h_l, 3 * D);
  }

  // --- readout ---
  votes_kernel<<<g1(NL * 32), 256, 0, stream>>>(h_l, out_w, out_b, (float*)d_out + BB);
  batch_reduce_kernel<<<dim3(BB), 256, 0, stream>>>((float*)d_out + BB, l_batch, (float*)d_out);
  copy_kernel<<<g1(NL * D), 256, 0, stream>>>(h_l, (float*)d_out + BB + NL, NL * D);
}
